// SelfAttention_74552042324501
// MI455X (gfx1250) — matrix operand order repaired and checked by reading, not compiled, not run on hardware
//
#include <hip/hip_runtime.h>

// Problem constants
#define B_  8
#define L_  1024
#define D_  768
#define H_  12
#define HD_ 64

typedef __attribute__((ext_vector_type(16))) __bf16 v16bf;
typedef __attribute__((ext_vector_type(8)))  __bf16 v8bf;
typedef __attribute__((ext_vector_type(8)))  float  v8f;
typedef __attribute__((ext_vector_type(4)))  int    v4i;

static __device__ __forceinline__ unsigned short f32_to_bf16(float f) {
  unsigned int u = __float_as_uint(f);
  u += 0x7FFFu + ((u >> 16) & 1u);   // round-to-nearest-even
  return (unsigned short)(u >> 16);
}

static __device__ __forceinline__ v8bf ld8(const unsigned short* p) {
  return *(const v8bf*)p;
}
static __device__ __forceinline__ v16bf ld16(const unsigned short* p) {
  return *(const v16bf*)p;
}
static __device__ __forceinline__ v16bf mk16(v8bf lo, v8bf hi) {
  union { v16bf v; v8bf h[2]; } u;
  u.h[0] = lo; u.h[1] = hi;
  return u.v;
}
static __device__ __forceinline__ v8f wmma_bf16(v16bf a, v16bf b, v8f c) {
  return __builtin_amdgcn_wmma_f32_16x16x32_bf16(false, a, false, b, (short)0, c,
                                                 false, false);
}

// --- CDNA5 async global->LDS copy (ASYNCcnt path) ------------------------
static __device__ __forceinline__ void async_load_b128(const void* gptr,
                                                       void* lptr) {
#if __has_builtin(__builtin_amdgcn_global_load_async_to_lds_b128)
  __builtin_amdgcn_global_load_async_to_lds_b128(
      (__attribute__((address_space(1))) v4i*)(v4i*)(void*)gptr,
      (__attribute__((address_space(3))) v4i*)(v4i*)lptr, 0, 0);
#else
  unsigned ldsoff = (unsigned)(unsigned long long)(
      (__attribute__((address_space(3))) void*)lptr);
  unsigned long long gaddr = (unsigned long long)gptr;
  asm volatile("global_load_async_to_lds_b128 %0, %1, off"
               :: "v"(ldsoff), "v"(gaddr) : "memory");
#endif
}
#define WAIT_ASYNC_0() asm volatile("s_wait_asynccnt 0x0" ::: "memory")
#define WAIT_ASYNC_1() asm volatile("s_wait_asynccnt 0x1" ::: "memory")

// ---------------------------------------------------------------------------
// Kernel 1: RMSNorm (fp32 in) -> bf16 xn.  One block per token row.
// ---------------------------------------------------------------------------
__global__ void rmsnorm_kernel(const float* __restrict__ x,
                               const float* __restrict__ nw,
                               unsigned short* __restrict__ xn) {
  int row = blockIdx.x;               // b*L + l
  int tid = threadIdx.x;              // 256 threads, 3 elems each
  const float* xr = x + (size_t)row * D_;
  float v0 = xr[tid], v1 = xr[tid + 256], v2 = xr[tid + 512];
  float ss = v0 * v0 + v1 * v1 + v2 * v2;
#pragma unroll
  for (int m = 16; m >= 1; m >>= 1) ss += __shfl_xor(ss, m, 32);
  __shared__ float wsum[8];
  int wave = tid >> 5, lane = tid & 31;
  if (lane == 0) wsum[wave] = ss;
  __syncthreads();
  float tot = 0.f;
#pragma unroll
  for (int w = 0; w < 8; ++w) tot += wsum[w];
  float r = rsqrtf(tot * (1.0f / D_) + 1.1920929e-7f);
  unsigned short* orow = xn + (size_t)row * D_;
  orow[tid]       = f32_to_bf16(v0 * r * nw[tid]);
  orow[tid + 256] = f32_to_bf16(v1 * r * nw[tid + 256]);
  orow[tid + 512] = f32_to_bf16(v2 * r * nw[tid + 512]);
}

// ---------------------------------------------------------------------------
// Kernel 2: fp32 -> bf16 weight cast
// ---------------------------------------------------------------------------
__global__ void cast_kernel(const float* __restrict__ src,
                            unsigned short* __restrict__ dst, int n) {
  int i = blockIdx.x * blockDim.x + threadIdx.x;
  if (i < n) dst[i] = f32_to_bf16(src[i]);
}

// ---------------------------------------------------------------------------
// GEMM building block: 8 waves/block; wave computes 16 rows x 64 cols.
// B panel (64 x 32 per k-step) staged in LDS via async DMA, double buffered.
// LDS layout: 64 rows (output cols) x 32 k, row padded to 40 ushorts (80 B).
// ---------------------------------------------------------------------------
#define BROW_PAD 40
#define KSTEPS   (D_ / 32)   // 24

// ---------------------------------------------------------------------------
// Kernel 3: QKV GEMM.  C[8192,2304] = xn @ wqkv^T + bqkv, scattered into
// q,k: [b,h,l,64] bf16 ; v stored transposed: [b,h,64,l] bf16.
// ---------------------------------------------------------------------------
__global__ void qkv_gemm_kernel(const unsigned short* __restrict__ xn,
                                const unsigned short* __restrict__ w,
                                const float* __restrict__ bias,
                                unsigned short* __restrict__ qg,
                                unsigned short* __restrict__ kg,
                                unsigned short* __restrict__ vg) {
  alignas(16) __shared__ unsigned short btile[2][64 * BROW_PAD];
  int tid = threadIdx.x, wave = tid >> 5, lane = tid & 31;
  int lane15 = lane & 15, hi = lane >> 4;
  int row0 = blockIdx.x * 128 + wave * 16;
  int col0 = blockIdx.y * 64;

  // per-thread async copy slot: 16 bytes of the 4 KB B tile
  int tcol = tid >> 2;            // 0..63  (output column within panel)
  int tk   = (tid & 3) * 8;       // 0,8,16,24 (k element offset)
  const unsigned short* gsrc = w + (size_t)(col0 + tcol) * D_ + tk;
  unsigned short* lds0 = &btile[0][tcol * BROW_PAD + tk];
  unsigned short* lds1 = &btile[1][tcol * BROW_PAD + tk];

  async_load_b128(gsrc, lds0);    // prefetch k-step 0

  v8f z = {0.f, 0.f, 0.f, 0.f, 0.f, 0.f, 0.f, 0.f};
  v8f acc[4] = {z, z, z, z};
  const unsigned short* arow = xn + (size_t)(row0 + lane15) * D_;

#pragma unroll
  for (int i = 0; i < KSTEPS; ++i) {
    int k0 = i * 32;
    if (i + 1 < KSTEPS) {
      async_load_b128(gsrc + k0 + 32, (i & 1) ? lds0 : lds1);
      WAIT_ASYNC_1();             // oldest (current tile) complete
    } else {
      WAIT_ASYNC_0();
    }
    __syncthreads();              // tile visible to all waves

    v16bf a = mk16(ld8(arow + k0 + hi * 16), ld8(arow + k0 + hi * 16 + 8));
    if (k0 + 32 < D_) __builtin_prefetch(arow + k0 + 32, 0, 1);
    const unsigned short* bb = &btile[i & 1][0];
#pragma unroll
    for (int t = 0; t < 4; ++t)
      acc[t] = wmma_bf16(a, ld16(bb + (t * 16 + lane15) * BROW_PAD + hi * 16),
                         acc[t]);
    __syncthreads();              // all reads done before buffer reuse
  }

  // Epilogue: add bias, scatter per head. col block is uniform in {q,k,v}.
#pragma unroll
  for (int t = 0; t < 4; ++t) {
    int mcol = col0 + t * 16 + lane15;
    int tsel = mcol / D_;
    int rem  = mcol - tsel * D_;
    int head = rem >> 6;
    int e    = rem & 63;
    float bv = bias[mcol];
#pragma unroll
    for (int r = 0; r < 8; ++r) {
      int n  = row0 + r + hi * 8;       // global token row
      int bb2 = n >> 10;
      int ll  = n & 1023;
      unsigned short val = f32_to_bf16(acc[t][r] + bv);
      if (tsel == 0)
        qg[((size_t)(bb2 * H_ + head) * L_ + ll) * HD_ + e] = val;
      else if (tsel == 1)
        kg[((size_t)(bb2 * H_ + head) * L_ + ll) * HD_ + e] = val;
      else
        vg[((size_t)(bb2 * H_ + head) * HD_ + e) * L_ + ll] = val;
    }
  }
}

// ---------------------------------------------------------------------------
// Kernel 4: flash attention with ALiBi-distance bias.
// grid = (L/128, H, B); block = 8 waves; wave owns 16 q rows, full HD=64.
// ---------------------------------------------------------------------------
__global__ void attn_kernel(const unsigned short* __restrict__ qg,
                            const unsigned short* __restrict__ kg,
                            const unsigned short* __restrict__ vg,
                            const float* __restrict__ pos,
                            const float* __restrict__ slopes,
                            unsigned short* __restrict__ og) {
  int qblk = blockIdx.x, h = blockIdx.y, b = blockIdx.z;
  int tid = threadIdx.x, wave = tid >> 5, lane = tid & 31;
  int lane15 = lane & 15, hi = lane >> 4;
  int qbase = qblk * 128 + wave * 16;

  size_t headoff = (size_t)(b * H_ + h) * L_ * HD_;
  const unsigned short* Qh = qg + headoff;
  const unsigned short* Kh = kg + headoff;
  const unsigned short* Vh = vg + headoff;  // [64][L]
  float slope = slopes[h];

  // Q A-fragments for the two 32-wide d-steps (fixed for whole kernel)
  const unsigned short* qrow = Qh + (size_t)(qbase + lane15) * HD_;
  v16bf qa0 = mk16(ld8(qrow + hi * 16),      ld8(qrow + hi * 16 + 8));
  v16bf qa1 = mk16(ld8(qrow + 32 + hi * 16), ld8(qrow + 32 + hi * 16 + 8));

  // q positions for this lane's 8 rows
  float qpx[8], qpy[8];
#pragma unroll
  for (int r = 0; r < 8; ++r) {
    int qi = qbase + hi * 8 + r;
    qpx[r] = pos[((size_t)b * L_ + qi) * 2];
    qpy[r] = pos[((size_t)b * L_ + qi) * 2 + 1];
  }

  float mrow[8], lrow[8];
  v8f z = {0.f, 0.f, 0.f, 0.f, 0.f, 0.f, 0.f, 0.f};
  v8f oacc[4] = {z, z, z, z};
#pragma unroll
  for (int r = 0; r < 8; ++r) { mrow[r] = -1e30f; lrow[r] = 0.f; }

  alignas(16) __shared__ unsigned short pbuf[8 * 16 * 32]; // per-wave P tile
  unsigned short* myp = pbuf + wave * 512;

  for (int kb = 0; kb < L_; kb += 32) {
    // --- scores for 32 keys: two 16x16 tiles ---
    v8f s0 = z, s1 = z;
    const unsigned short* k0p = Kh + (size_t)(kb + lane15) * HD_ + hi * 16;
    s0 = wmma_bf16(qa0, ld16(k0p), s0);
    s0 = wmma_bf16(qa1, ld16(k0p + 32), s0);
    const unsigned short* k1p = Kh + (size_t)(kb + 16 + lane15) * HD_ + hi * 16;
    s1 = wmma_bf16(qa0, ld16(k1p), s1);
    s1 = wmma_bf16(qa1, ld16(k1p + 32), s1);

    // --- scale + ALiBi distance bias ---
    float kx0 = pos[((size_t)b * L_ + kb + lane15) * 2];
    float ky0 = pos[((size_t)b * L_ + kb + lane15) * 2 + 1];
    float kx1 = pos[((size_t)b * L_ + kb + 16 + lane15) * 2];
    float ky1 = pos[((size_t)b * L_ + kb + 16 + lane15) * 2 + 1];
#pragma unroll
    for (int r = 0; r < 8; ++r) {
      float dx0 = qpx[r] - kx0, dy0 = qpy[r] - ky0;
      float dx1 = qpx[r] - kx1, dy1 = qpy[r] - ky1;
      s0[r] = s0[r] * 0.125f + slope * sqrtf(dx0 * dx0 + dy0 * dy0);
      s1[r] = s1[r] * 0.125f + slope * sqrtf(dx1 * dx1 + dy1 * dy1);
    }

    // --- online softmax ---
    float cm[8];
#pragma unroll
    for (int r = 0; r < 8; ++r) cm[r] = fmaxf(s0[r], s1[r]);
#pragma unroll
    for (int m = 1; m <= 8; m <<= 1)
#pragma unroll
      for (int r = 0; r < 8; ++r) cm[r] = fmaxf(cm[r], __shfl_xor(cm[r], m, 32));

    float alpha[8], p0[8], p1[8], rs[8];
#pragma unroll
    for (int r = 0; r < 8; ++r) {
      float mn = fmaxf(mrow[r], cm[r]);
      alpha[r] = __expf(mrow[r] - mn);
      mrow[r] = mn;
      p0[r] = __expf(s0[r] - mn);
      p1[r] = __expf(s1[r] - mn);
      rs[r] = p0[r] + p1[r];
    }
#pragma unroll
    for (int m = 1; m <= 8; m <<= 1)
#pragma unroll
      for (int r = 0; r < 8; ++r) rs[r] += __shfl_xor(rs[r], m, 32);
#pragma unroll
    for (int r = 0; r < 8; ++r) lrow[r] = lrow[r] * alpha[r] + rs[r];
#pragma unroll
    for (int t = 0; t < 4; ++t)
#pragma unroll
      for (int r = 0; r < 8; ++r) oacc[t][r] *= alpha[r];

    // --- P: C-layout -> LDS -> A-fragment (wave-private region) ---
#pragma unroll
    for (int r = 0; r < 8; ++r) {
      myp[(r + hi * 8) * 32 + lane15]      = f32_to_bf16(p0[r]);
      myp[(r + hi * 8) * 32 + 16 + lane15] = f32_to_bf16(p1[r]);
    }
    const unsigned short* pr = myp + lane15 * 32 + hi * 16;
    v16bf pa = mk16(ld8(pr), ld8(pr + 8));

    // --- O += P x V (V transposed: keys contiguous) ---
#pragma unroll
    for (int t = 0; t < 4; ++t) {
      const unsigned short* vp =
          Vh + (size_t)(t * 16 + lane15) * L_ + kb + hi * 16;
      oacc[t] = wmma_bf16(pa, ld16(vp), oacc[t]);
    }
  }

  // --- normalize and store o as bf16 [b*L, 768] ---
  float inv[8];
#pragma unroll
  for (int r = 0; r < 8; ++r) inv[r] = 1.0f / lrow[r];
#pragma unroll
  for (int t = 0; t < 4; ++t)
#pragma unroll
    for (int r = 0; r < 8; ++r) {
      int row = qbase + r + hi * 8;
      int col = h * HD_ + t * 16 + lane15;
      og[(size_t)(b * L_ + row) * D_ + col] = f32_to_bf16(oacc[t][r] * inv[r]);
    }
}

// ---------------------------------------------------------------------------
// Kernel 5: output projection.  out[8192,768] = o @ wout^T + bout (fp32 out)
// ---------------------------------------------------------------------------
__global__ void out_gemm_kernel(const unsigned short* __restrict__ og,
                                const unsigned short* __restrict__ w,
                                const float* __restrict__ bias,
                                float* __restrict__ out) {
  alignas(16) __shared__ unsigned short btile[2][64 * BROW_PAD];
  int tid = threadIdx.x, wave = tid >> 5, lane = tid & 31;
  int lane15 = lane & 15, hi = lane >> 4;
  int row0 = blockIdx.x * 128 + wave * 16;
  int col0 = blockIdx.y * 64;

  int tcol = tid >> 2;
  int tk   = (tid & 3) * 8;
  const unsigned short* gsrc = w + (size_t)(col0 + tcol) * D_ + tk;
  unsigned short* lds0 = &btile[0][tcol * BROW_PAD + tk];
  unsigned short* lds1 = &btile[1][tcol * BROW_PAD + tk];

  async_load_b128(gsrc, lds0);

  v8f z = {0.f, 0.f, 0.f, 0.f, 0.f, 0.f, 0.f, 0.f};
  v8f acc[4] = {z, z, z, z};
  const unsigned short* arow = og + (size_t)(row0 + lane15) * D_;

#pragma unroll
  for (int i = 0; i < KSTEPS; ++i) {
    int k0 = i * 32;
    if (i + 1 < KSTEPS) {
      async_load_b128(gsrc + k0 + 32, (i & 1) ? lds0 : lds1);
      WAIT_ASYNC_1();
    } else {
      WAIT_ASYNC_0();
    }
    __syncthreads();

    v16bf a = mk16(ld8(arow + k0 + hi * 16), ld8(arow + k0 + hi * 16 + 8));
    if (k0 + 32 < D_) __builtin_prefetch(arow + k0 + 32, 0, 1);
    const unsigned short* bb = &btile[i & 1][0];
#pragma unroll
    for (int t = 0; t < 4; ++t)
      acc[t] = wmma_bf16(a, ld16(bb + (t * 16 + lane15) * BROW_PAD + hi * 16),
                         acc[t]);
    __syncthreads();
  }

#pragma unroll
  for (int t = 0; t < 4; ++t) {
    int mcol = col0 + t * 16 + lane15;
    float bv = bias[mcol];
#pragma unroll
    for (int r = 0; r < 8; ++r) {
      int n = row0 + r + hi * 8;
      out[(size_t)n * D_ + mcol] = acc[t][r] + bv;
    }
  }
}

// ---------------------------------------------------------------------------
extern "C" void kernel_launch(void* const* d_in, const int* in_sizes, int n_in,
                              void* d_out, int out_size, void* d_ws,
                              size_t ws_size, hipStream_t stream) {
  (void)in_sizes; (void)n_in; (void)out_size; (void)ws_size;
  const float* x      = (const float*)d_in[0];
  const float* pos    = (const float*)d_in[1];
  const float* wqkv   = (const float*)d_in[2];
  const float* bqkv   = (const float*)d_in[3];
  const float* nw     = (const float*)d_in[4];
  const float* wout   = (const float*)d_in[5];
  const float* bout   = (const float*)d_in[6];
  const float* slopes = (const float*)d_in[7];
  float* out = (float*)d_out;

  char* ws = (char*)d_ws;
  // bf16 workspace layout (bytes)
  unsigned short* xn = (unsigned short*)(ws);                         // 12,582,912
  unsigned short* wq = (unsigned short*)(ws + 12582912);              //  3,538,944
  unsigned short* wo = (unsigned short*)(ws + 16121856);              //  1,179,648
  unsigned short* qg = (unsigned short*)(ws + 17301504);              // 12,582,912
  unsigned short* kg = (unsigned short*)(ws + 29884416);              // 12,582,912
  unsigned short* vg = (unsigned short*)(ws + 42467328);              // 12,582,912
  unsigned short* og = (unsigned short*)(ws + 55050240);              // 12,582,912
                                                                      // total ~67.6 MB
  rmsnorm_kernel<<<B_ * L_, 256, 0, stream>>>(x, nw, xn);
  cast_kernel<<<(3 * D_ * D_ + 255) / 256, 256, 0, stream>>>(wqkv, wq, 3 * D_ * D_);
  cast_kernel<<<(D_ * D_ + 255) / 256, 256, 0, stream>>>(wout, wo, D_ * D_);
  qkv_gemm_kernel<<<dim3(64, 36), 256, 0, stream>>>(xn, wq, bqkv, qg, kg, vg);
  attn_kernel<<<dim3(L_ / 128, H_, B_), 256, 0, stream>>>(qg, kg, vg, pos, slopes, og);
  out_gemm_kernel<<<dim3(64, 12), 256, 0, stream>>>(og, wo, bout, out);
}
